// TinyTransformerBlock_81673098100996
// MI455X (gfx1250) — matrix operand order repaired and checked by reading, not compiled, not run on hardware
//
#include <hip/hip_runtime.h>
#include <hip/hip_bf16.h>

typedef __bf16 bf16;
typedef __attribute__((ext_vector_type(2)))  __bf16 v2bf;
typedef __attribute__((ext_vector_type(8)))  __bf16 v8bf;
typedef __attribute__((ext_vector_type(16))) __bf16 v16bf;
typedef __attribute__((ext_vector_type(8)))  float  v8f;

// ---------------------------------------------------------------------------
// fp32 -> bf16 conversion, 8-wide (all tensor sizes are multiples of 2048)
// ---------------------------------------------------------------------------
__global__ __launch_bounds__(256) void k_cvt(const float* __restrict__ s,
                                             bf16* __restrict__ d, long n) {
  long i = ((long)blockIdx.x * blockDim.x + threadIdx.x) * 8;
  if (i >= n) return;
  const float4* p = (const float4*)(s + i);
  float4 a = p[0], b = p[1];
  v8bf o;
  o[0] = (bf16)a.x; o[1] = (bf16)a.y; o[2] = (bf16)a.z; o[3] = (bf16)a.w;
  o[4] = (bf16)b.x; o[5] = (bf16)b.y; o[6] = (bf16)b.z; o[7] = (bf16)b.w;
  *(v8bf*)(d + i) = o;
}

// ---------------------------------------------------------------------------
// Generic bf16 WMMA GEMM: C[z] = scale * A[z] @ B[z] (+bias) (+epilogues)
//   BM=BN=128, BK=64, 256 threads = 8 waves (4 x 2), wave tile 32x64,
//   16 x v_wmma_f32_16x16x32_bf16 per barrier.
//   BT:       B stored [N,K] row-major (used for Q @ K^T)
//   RELU:     relu after bias
//   GATE:     multiply by gates[row*8 + blockIdx.z] (MoE stage-1 scaling)
//   MOEBIAS:  add sum_e gates[row*8+e] * b2[e*N + col] (MoE stage-2)
//   ADDRES:   add res[row*ldcf + col]
// All M,N divisible by 128, K divisible by 64 (true for every call here).
// ---------------------------------------------------------------------------
template<bool BT, bool RELU, bool GATE, bool MOEBIAS, bool ADDRES>
__global__ __launch_bounds__(256) void k_gemm(
    const bf16* __restrict__ A, const bf16* __restrict__ Bm,
    const float* __restrict__ bias,
    float* __restrict__ outF, bf16* __restrict__ outB,
    const float* __restrict__ res,
    const float* __restrict__ gates, const float* __restrict__ b2,
    int M, int N, int K, int lda, int ldb, int ldcf, int ldcb,
    long sA, long sB, long sCf, long sCb, long sBias, float scale) {
  constexpr int BM = 128, BN = 128, BK = 64, BKp = BK + 8;  // 144B rows: 16B aligned
  __shared__ bf16 As[BM][BKp];
  __shared__ bf16 Bs[BN][BKp];  // stored [n][k] for contiguous B-fragment reads

  const int tid  = threadIdx.x;
  const int lane = tid & 31, wid = tid >> 5;
  const int wm = (wid & 3) * 32;   // wave row offset in tile
  const int wn = (wid >> 2) * 64;  // wave col offset in tile
  const int m0 = blockIdx.y * BM, n0 = blockIdx.x * BN;
  const int z  = blockIdx.z;
  const int half = lane >> 4, l15 = lane & 15;

  A  += (long)z * sA;
  Bm += (long)z * sB;

  v8f acc[2][4];
#pragma unroll
  for (int s = 0; s < 2; ++s)
#pragma unroll
    for (int t = 0; t < 4; ++t) acc[s][t] = (v8f){0,0,0,0,0,0,0,0};

  for (int k0 = 0; k0 < K; k0 += BK) {
    __syncthreads();
    // ---- stage A tile (128x64): 1024 x 16B, 4 per thread ----
#pragma unroll
    for (int it = 0; it < 4; ++it) {
      int idx = tid + it * 256;
      int r = idx >> 3, seg = (idx & 7) * 8;
      *(v8bf*)&As[r][seg] = *(const v8bf*)&A[(long)(m0 + r) * lda + k0 + seg];
    }
    // ---- stage B tile into [n][k] layout ----
    if (BT) {
#pragma unroll
      for (int it = 0; it < 4; ++it) {
        int idx = tid + it * 256;
        int r = idx >> 3, seg = (idx & 7) * 8;
        *(v8bf*)&Bs[r][seg] = *(const v8bf*)&Bm[(long)(n0 + r) * ldb + k0 + seg];
      }
    } else {
      // two consecutive k-rows per unit -> packed 2xbf16 b32 LDS stores
#pragma unroll
      for (int it = 0; it < 2; ++it) {
        int u = tid + it * 256;
        int kp = u >> 4, nseg = (u & 15) * 8;  // kp in [0,32): k pair index
        v8bf b0 = *(const v8bf*)&Bm[(long)(k0 + 2 * kp + 0) * ldb + n0 + nseg];
        v8bf b1 = *(const v8bf*)&Bm[(long)(k0 + 2 * kp + 1) * ldb + n0 + nseg];
#pragma unroll
        for (int j = 0; j < 8; ++j)
          *(v2bf*)&Bs[nseg + j][2 * kp] = (v2bf){b0[j], b1[j]};
      }
    }
    // ---- prefetch next K-slab into L2 while this one is consumed ----
    if (k0 + BK < K) {
      {
        int r = tid >> 3, seg = (tid & 7) * 8;
        __builtin_prefetch(&A[(long)(m0 + r) * lda + k0 + BK + seg], 0, 1);
      }
      if (BT) {
        int r = tid >> 3, seg = (tid & 7) * 8;
        __builtin_prefetch(&Bm[(long)(n0 + r) * ldb + k0 + BK + seg], 0, 1);
      } else {
        int kp = tid >> 4, nseg = (tid & 15) * 8;
        __builtin_prefetch(&Bm[(long)(k0 + BK + 2 * kp) * ldb + n0 + nseg], 0, 1);
      }
    }
    __syncthreads();

    // ---- 2 x k-step of 32, 8 WMMA each ----
#pragma unroll
    for (int kc = 0; kc < BK; kc += 32) {
      // A 16x32: lane half h holds K = {8h..8h+7, 16+8h..16+8h+7} for row l15
      v16bf af[2];
#pragma unroll
      for (int s = 0; s < 2; ++s) {
        const bf16* p = &As[wm + s * 16 + l15][kc + 8 * half];
        v8bf lo = *(const v8bf*)p;
        v8bf hi = *(const v8bf*)(p + 16);
        af[s] = __builtin_shufflevector(lo, hi, 0,1,2,3,4,5,6,7,8,9,10,11,12,13,14,15);
      }
      // B 32x16: lane half h holds K = {8h..8h+7, 16+8h..16+8h+7} for col l15
      v16bf bfr[4];
#pragma unroll
      for (int t = 0; t < 4; ++t) {
        const bf16* p = &Bs[wn + t * 16 + l15][kc + 8 * half];
        v8bf lo = *(const v8bf*)p;
        v8bf hi = *(const v8bf*)(p + 16);
        bfr[t] = __builtin_shufflevector(lo, hi, 0,1,2,3,4,5,6,7,8,9,10,11,12,13,14,15);
      }
#pragma unroll
      for (int s = 0; s < 2; ++s)
#pragma unroll
        for (int t = 0; t < 4; ++t)
          acc[s][t] = __builtin_amdgcn_wmma_f32_16x16x32_bf16(
              false, af[s], false, bfr[t], (short)0, acc[s][t], false, false);
    }
  }

  // ---- epilogue: D layout lane half h, VGPR r -> row = r + 8h, col = l15 ----
#pragma unroll
  for (int s = 0; s < 2; ++s) {
#pragma unroll
    for (int t = 0; t < 4; ++t) {
      const int col = n0 + wn + t * 16 + l15;
      float bcol = bias ? bias[(long)z * sBias + col] : 0.0f;
      float mb = 0.0f;
      (void)mb;
#pragma unroll
      for (int r = 0; r < 8; ++r) {
        const int row = m0 + wm + s * 16 + half * 8 + r;
        float c = acc[s][t][r] * scale + bcol;
        if (RELU) c = fmaxf(c, 0.0f);
        if (GATE) c *= gates[(long)row * 8 + z];
        if (MOEBIAS) {
          const float* g = &gates[(long)row * 8];
#pragma unroll
          for (int e = 0; e < 8; ++e) c += g[e] * b2[(long)e * N + col];
        }
        if (ADDRES) c += res[(long)row * ldcf + col];
        if (outF) outF[(long)z * sCf + (long)row * ldcf + col] = c;
        if (outB) outB[(long)z * sCb + (long)row * ldcb + col] = (bf16)c;
      }
    }
  }
}

// ---------------------------------------------------------------------------
// Row softmax (row length = 1024) fp32 in -> bf16 out. One block per row.
// ---------------------------------------------------------------------------
__global__ __launch_bounds__(256) void k_softmax_row(const float* __restrict__ s,
                                                     bf16* __restrict__ attn,
                                                     int ncol) {
  __shared__ float redm[8];
  __shared__ float reds[8];
  const long row = blockIdx.x;
  const float* p = s + row * (long)ncol;
  const int tid = threadIdx.x, lane = tid & 31, wid = tid >> 5;

  float lmax = -1e30f;
  for (int c = tid; c < ncol; c += 256) lmax = fmaxf(lmax, p[c]);
#pragma unroll
  for (int o = 16; o > 0; o >>= 1) lmax = fmaxf(lmax, __shfl_xor(lmax, o, 32));
  if (lane == 0) redm[wid] = lmax;
  __syncthreads();
  float m = redm[0];
#pragma unroll
  for (int i = 1; i < 8; ++i) m = fmaxf(m, redm[i]);

  float lsum = 0.0f;
  for (int c = tid; c < ncol; c += 256) lsum += __expf(p[c] - m);
#pragma unroll
  for (int o = 16; o > 0; o >>= 1) lsum += __shfl_xor(lsum, o, 32);
  if (lane == 0) reds[wid] = lsum;
  __syncthreads();
  float ssum = 0.0f;
#pragma unroll
  for (int i = 0; i < 8; ++i) ssum += reds[i];
  const float inv = 1.0f / ssum;

  bf16* q = attn + row * (long)ncol;
  for (int c = tid; c < ncol; c += 256) q[c] = (bf16)(__expf(p[c] - m) * inv);
}

// ---------------------------------------------------------------------------
// Gating: gates = softmax(x1 @ Wg + bg) over E=8. One wave per token row.
// ---------------------------------------------------------------------------
__global__ __launch_bounds__(128) void k_gates(const float* __restrict__ x1,
                                               const float* __restrict__ Wg,
                                               const float* __restrict__ bg,
                                               float* __restrict__ gates, int Hd) {
  const int lane = threadIdx.x & 31, w = threadIdx.x >> 5;
  const long row = (long)blockIdx.x * 4 + w;
  const float* xp = x1 + row * Hd;
  float acc[8] = {0, 0, 0, 0, 0, 0, 0, 0};
  for (int k = lane; k < Hd; k += 32) {
    const float xv = xp[k];
    const float* wr = Wg + (long)k * 8;
#pragma unroll
    for (int e = 0; e < 8; ++e) acc[e] += xv * wr[e];
  }
#pragma unroll
  for (int e = 0; e < 8; ++e)
#pragma unroll
    for (int o = 16; o > 0; o >>= 1) acc[e] += __shfl_xor(acc[e], o, 32);
  if (lane == 0) {
    float mx = -1e30f;
#pragma unroll
    for (int e = 0; e < 8; ++e) { acc[e] += bg[e]; mx = fmaxf(mx, acc[e]); }
    float s = 0.0f;
#pragma unroll
    for (int e = 0; e < 8; ++e) { acc[e] = __expf(acc[e] - mx); s += acc[e]; }
    const float inv = 1.0f / s;
    float* gp = gates + row * 8;
#pragma unroll
    for (int e = 0; e < 8; ++e) gp[e] = acc[e] * inv;
  }
}

// ---------------------------------------------------------------------------
extern "C" void kernel_launch(void* const* d_in, const int* in_sizes, int n_in,
                              void* d_out, int out_size, void* d_ws, size_t ws_size,
                              hipStream_t stream) {
  (void)in_sizes; (void)n_in; (void)out_size; (void)ws_size;
  constexpr int Bn = 4, S = 1024, H = 1024, D = 4096, E = 8;
  constexpr int M = Bn * S;                 // 4096 tokens
  constexpr long KD = (long)E * D;          // 32768

  const float* x  = (const float*)d_in[0];
  const float* Wq = (const float*)d_in[1];  const float* bq = (const float*)d_in[2];
  const float* Wk = (const float*)d_in[3];  const float* bk = (const float*)d_in[4];
  const float* Wv = (const float*)d_in[5];  const float* bv = (const float*)d_in[6];
  const float* Wo = (const float*)d_in[7];  const float* bo = (const float*)d_in[8];
  const float* Wg = (const float*)d_in[9];  const float* bg = (const float*)d_in[10];
  const float* W1 = (const float*)d_in[11]; const float* b1 = (const float*)d_in[12];
  const float* W2 = (const float*)d_in[13]; const float* b2 = (const float*)d_in[14];
  float* out = (float*)d_out;

  // workspace carve-up
  char* w = (char*)d_ws;
  auto take = [&](size_t bytes) {
    char* p = w;
    w += (bytes + 255) & ~(size_t)255;
    return p;
  };
  bf16*  xbf  = (bf16*) take((size_t)M * H * 2);
  bf16*  wqb  = (bf16*) take((size_t)H * H * 2);
  bf16*  wkb  = (bf16*) take((size_t)H * H * 2);
  bf16*  wvb  = (bf16*) take((size_t)H * H * 2);
  bf16*  wob  = (bf16*) take((size_t)H * H * 2);
  bf16*  qb   = (bf16*) take((size_t)M * H * 2);
  bf16*  kb   = (bf16*) take((size_t)M * H * 2);
  bf16*  vb   = (bf16*) take((size_t)M * H * 2);
  float* sc   = (float*)take((size_t)Bn * S * S * 4);
  bf16*  atb  = (bf16*) take((size_t)Bn * S * S * 2);
  bf16*  avb  = (bf16*) take((size_t)M * H * 2);
  float* x1   = (float*)take((size_t)M * H * 4);
  bf16*  x1b  = (bf16*) take((size_t)M * H * 2);
  float* gts  = (float*)take((size_t)M * E * 4);
  bf16*  w1b  = (bf16*) take((size_t)E * H * D * 2);
  bf16*  w2b  = (bf16*) take((size_t)E * D * H * 2);
  bf16*  hm   = (bf16*) take((size_t)M * KD * 2);   // gate-scaled hmid [M, E*D]

  auto cvt = [&](const float* s, bf16* d, long n) {
    k_cvt<<<dim3((unsigned)(n / (8 * 256))), dim3(256), 0, stream>>>(s, d, n);
  };
  cvt(x,  xbf, (long)M * H);
  cvt(Wq, wqb, (long)H * H);
  cvt(Wk, wkb, (long)H * H);
  cvt(Wv, wvb, (long)H * H);
  cvt(Wo, wob, (long)H * H);
  cvt(W1, w1b, (long)E * H * D);
  cvt(W2, w2b, (long)E * D * H);

  const dim3 blk(256);

  // ---- Q, K, V projections (bf16 out) ----
  auto proj = [&](const bf16* Wb, const float* bvec, bf16* o) {
    k_gemm<false, false, false, false, false>
        <<<dim3(H / 128, M / 128, 1), blk, 0, stream>>>(
            xbf, Wb, bvec, nullptr, o, nullptr, nullptr, nullptr,
            M, H, H, H, H, H, H, 0, 0, 0, 0, 0, 1.0f);
  };
  proj(wqb, bq, qb);
  proj(wkb, bk, kb);
  proj(wvb, bv, vb);

  // ---- scores = (q @ k^T) / sqrt(H), per batch ----
  k_gemm<true, false, false, false, false>
      <<<dim3(S / 128, S / 128, Bn), blk, 0, stream>>>(
          qb, kb, nullptr, sc, nullptr, nullptr, nullptr, nullptr,
          S, S, H, H, H, S, S,
          (long)S * H, (long)S * H, (long)S * S, 0, 0, 1.0f / 32.0f);

  // ---- softmax rows -> bf16 attn ----
  k_softmax_row<<<dim3(Bn * S), blk, 0, stream>>>(sc, atb, S);

  // ---- a = attn @ v (bf16 out), per batch ----
  k_gemm<false, false, false, false, false>
      <<<dim3(H / 128, S / 128, Bn), blk, 0, stream>>>(
          atb, vb, nullptr, nullptr, avb, nullptr, nullptr, nullptr,
          S, H, S, S, H, H, H,
          (long)S * S, (long)S * H, 0, (long)S * H, 0, 1.0f);

  // ---- x1 = x + a @ Wo + bo (dual f32 + bf16 out) ----
  k_gemm<false, false, false, false, true>
      <<<dim3(H / 128, M / 128, 1), blk, 0, stream>>>(
          avb, wob, bo, x1, x1b, x, nullptr, nullptr,
          M, H, H, H, H, H, H, 0, 0, 0, 0, 0, 1.0f);

  // ---- gates = softmax(x1 @ Wg + bg) ----
  k_gates<<<dim3(M / 4), dim3(128), 0, stream>>>(x1, Wg, bg, gts, H);

  // ---- stage 1: hm[:, e*D:(e+1)*D] = gates[:,e] * relu(x1 @ W1[e] + b1[e]) ----
  k_gemm<false, true, true, false, false>
      <<<dim3(D / 128, M / 128, E), blk, 0, stream>>>(
          x1b, w1b, b1, nullptr, hm, nullptr, gts, nullptr,
          M, D, H, H, D, D, (int)KD,
          0, (long)H * D, 0, (long)D, (long)D, 1.0f);

  // ---- stage 2: out = x1 + hm @ W2_flat + sum_e gates_e * b2[e] ----
  k_gemm<false, false, false, true, true>
      <<<dim3(H / 128, M / 128, 1), blk, 0, stream>>>(
          hm, w2b, nullptr, out, nullptr, x1, gts, b2,
          M, H, (int)KD, (int)KD, H, H, H, 0, 0, 0, 0, 0, 1.0f);
}
